// continuous_convolution_network_56839597195964
// MI455X (gfx1250) — matrix operand order repaired and checked by reading, not compiled, not run on hardware
//
#include <hip/hip_runtime.h>
#include <hip/hip_bf16.h>

typedef _Float16 v16h __attribute__((ext_vector_type(16)));
typedef _Float16 h8   __attribute__((ext_vector_type(8)));
typedef float    v8f  __attribute__((ext_vector_type(8)));

#define HH 256
#define WW 256
#define BB 2
#define CCH 5
#define NPIX (BB*HH*WW)      // 131072
#define KNBk 5
#define NKROW (NPIX*KNBk)    // 655360
#define EPSBN 1e-5f

// -------- stats float layout (in ws) --------
// [0] cnt ; [16..25] vfe_sum ; [32..41] vfe_ss ;
// [48..175] mlp_sum ; [176..303] mlp_ss ; [304..308] post_sum ; [320..324] post_ss
// -------- par float layout --------
// [0..9] vfe_scale ; [16..25] vfe_shift ; [32..159] mlp_scale ; [160..287] mlp_shift ;
// [288..292] post_scale ; [304..308] post_shift

__device__ __forceinline__ v8f wmma16(v16h a, v16h b, v8f c) {
  return __builtin_amdgcn_wmma_f32_16x16x32_f16(false, a, false, b, (short)0, c, false, false);
}

// A fragment (16x32 f16), row = lane&15, data at rowbase[K] contiguous.
// lanes 0-15: K 0..15; lanes 16-31: K 16..31
__device__ __forceinline__ v16h load_afrag(const _Float16* rowbase, int lane, int ks) {
  int off = (lane >> 4) << 4; // 0 or 16
  h8 lo = *(const h8*)(rowbase + ks * 32 + off);
  h8 hi = *(const h8*)(rowbase + ks * 32 + 8 + off);
  return __builtin_shufflevector(lo, hi, 0,1,2,3,4,5,6,7,8,9,10,11,12,13,14,15);
}

// B fragment (32x16 f16): column nn contiguous K (stride halves), lane col = nn,
// lanes 0-15 hold K 0..15 of the k-step, lanes 16-31 hold K 16..31.
__device__ __forceinline__ v16h load_bfrag(const _Float16* B, int stride, int nn, int lane, int ks) {
  return *(const v16h*)(B + nn * stride + ks * 32 + ((lane >> 4) << 4));
}

// ---------------- kernel 0: zero stats ----------------
__global__ void k_zero(float* __restrict__ p, int n) {
  int i = blockIdx.x * 256 + threadIdx.x;
  if (i < n) p[i] = 0.f;
}

// ---------------- kernel 1: unfold + select + gather + vfe fc1 stats ------
__global__ __launch_bounds__(256) void k_prep(
    const float* __restrict__ in, const float* __restrict__ w1, const float* __restrict__ b1,
    float* __restrict__ coords, float* __restrict__ maskf, float* __restrict__ stats) {
  __shared__ float sS[10], sQ[10], sC;
  int t = threadIdx.x;
  if (t < 10) { sS[t] = 0.f; sQ[t] = 0.f; }
  if (t == 0) sC = 0.f;
  __syncthreads();

  int n = blockIdx.x * 256 + t;
  int bi = n >> 16, rem = n & 65535, y = rem >> 8, x = rem & 255;

  float uf[CCH][9];
  for (int ch = 0; ch < CCH; ++ch)
    for (int dy = 0; dy < 3; ++dy)
      for (int dx = 0; dx < 3; ++dx) {
        int yy = y + dy - 1, xx = x + dx - 1;
        float v = 0.f;
        if (yy >= 0 && yy < HH && xx >= 0 && xx < WW)
          v = in[(((size_t)bi * CCH + ch) * HH + yy) * WW + xx];
        uf[ch][dy * 3 + dx] = v;
      }
  float msum = 0.f;
  for (int j = 0; j < 9; ++j) msum += uf[0][j];
  float m = (msum > 0.f) ? 1.f : 0.f;

  // inverse gaussian (k=3, sigma=10); constant factor cancels in normalization
  float g[9], gs = 0.f;
  for (int dy = 0; dy < 3; ++dy)
    for (int dx = 0; dx < 3; ++dx) {
      float e = __expf(-(float)((dx - 1) * (dx - 1) + (dy - 1) * (dy - 1)) / 200.f);
      g[dy * 3 + dx] = e; gs += e;
    }
  float dist[9];
  for (int j = 0; j < 9; ++j) {
    float invg = 1.f - g[j] / gs;
    dist[j] = (fabsf(uf[0][j] - uf[0][4]) + fabsf(uf[CCH - 1][j] - uf[CCH - 1][4]) + 0.01f) * invg;
  }
  // stable 5 smallest (tie -> lowest index), ascending
  int sel[KNBk]; bool used[9];
  for (int j = 0; j < 9; ++j) used[j] = false;
  for (int s = 0; s < KNBk; ++s) {
    int best = -1;
    for (int j = 0; j < 9; ++j)
      if (!used[j] && (best < 0 || dist[j] < dist[best])) best = j;
    used[best] = true; sel[s] = best;
  }

  float ls[10], lq[10];
  for (int o = 0; o < 10; ++o) { ls[o] = 0.f; lq[o] = 0.f; }
  for (int s = 0; s < KNBk; ++s) {
    int j = sel[s];
    for (int ch = 0; ch < CCH; ++ch) coords[(size_t)n * 25 + ch * KNBk + s] = uf[ch][j];
    for (int o = 0; o < 10; ++o) {
      float h = b1[o];
      for (int ch = 0; ch < CCH; ++ch) h += w1[o * CCH + ch] * uf[ch][j];
      ls[o] += h; lq[o] += h * h;
    }
  }
  maskf[n] = m;
  if (m > 0.f) {
    for (int o = 0; o < 10; ++o) { atomicAdd(&sS[o], ls[o]); atomicAdd(&sQ[o], lq[o]); }
    atomicAdd(&sC, 1.f);
  }
  __syncthreads();
  if (t < 10) { atomicAdd(&stats[16 + t], sS[t]); atomicAdd(&stats[32 + t], sQ[t]); }
  if (t == 0) atomicAdd(&stats[0], sC);
}

// ---------------- finalize BN stats -> scale/shift ----------------
__global__ void k_finalize(const float* __restrict__ stats, float* __restrict__ par,
                           const float* __restrict__ gamma, const float* __restrict__ beta,
                           int so_sum, int so_ss, int po_sc, int po_sh, int nf, float cntScale) {
  int t = threadIdx.x;
  if (t < nf) {
    float c = stats[0] * cntScale;
    float mean = stats[so_sum + t] / c;
    float var = stats[so_ss + t] / c - mean * mean;
    float rstd = rsqrtf(var + EPSBN);
    float sc = rstd * gamma[t];
    par[po_sc + t] = sc;
    par[po_sh + t] = beta[t] - mean * sc;
  }
}

// ---------------- kernel 2: vfe apply + max/mean -> f60 (f16, K padded 64) --
__global__ __launch_bounds__(256) void k_vfe2(
    const float* __restrict__ coords,
    const float* __restrict__ w1, const float* __restrict__ b1,
    const float* __restrict__ par,
    const float* __restrict__ w2, const float* __restrict__ b2,
    _Float16* __restrict__ f60) {
  int n = blockIdx.x * 256 + threadIdx.x;
  float f[KNBk][20], mx[20], av[20];
  for (int o = 0; o < 20; ++o) { mx[o] = -1e30f; av[o] = 0.f; }
  for (int s = 0; s < KNBk; ++s) {
    float c5[CCH];
    for (int ch = 0; ch < CCH; ++ch) c5[ch] = coords[(size_t)n * 25 + ch * KNBk + s];
    float h[10];
    for (int o = 0; o < 10; ++o) {
      float a = b1[o];
      for (int ch = 0; ch < CCH; ++ch) a += w1[o * CCH + ch] * c5[ch];
      a = a * par[o] + par[16 + o];
      h[o] = fmaxf(a, 0.f);
    }
    for (int o = 0; o < 20; ++o) {
      float a = b2[o];
      for (int i = 0; i < 10; ++i) a += w2[o * 10 + i] * h[i];
      f[s][o] = a;
      mx[o] = fmaxf(mx[o], a);
      av[o] += a;
    }
  }
  for (int o = 0; o < 20; ++o) av[o] *= 0.2f;
  for (int s = 0; s < KNBk; ++s) {
    _Float16 row[64];
    for (int o = 0; o < 20; ++o) {
      row[o] = (_Float16)f[s][o];
      row[20 + o] = (_Float16)mx[o];
      row[40 + o] = (_Float16)av[o];
    }
    for (int o = 60; o < 64; ++o) row[o] = (_Float16)0.f;
    h8* dst = (h8*)(f60 + (size_t)(n * KNBk + s) * 64);
    const h8* src = (const h8*)row;
    for (int i = 0; i < 8; ++i) dst[i] = src[i];
  }
}

// ---------------- kernel 3: mlp fc1 (WMMA) stats pass ----------------
__global__ __launch_bounds__(256) void k_mlp_stats(
    const _Float16* __restrict__ f60, const float* __restrict__ maskf,
    const float* __restrict__ w7, const float* __restrict__ b8,
    float* __restrict__ stats) {
  __shared__ __align__(32) _Float16 Bl[128 * 64];
  __shared__ float sS[128], sQ[128];
  int t = threadIdx.x;
  for (int i = t; i < 128 * 64; i += 256) {
    int nn = i >> 6, kk = i & 63;
    Bl[i] = (_Float16)((nn < 120 && kk < 60) ? w7[nn * 60 + kk] : 0.f);
  }
  if (t < 128) { sS[t] = 0.f; sQ[t] = 0.f; }
  __syncthreads();

  int wave = t >> 5, lane = t & 31;
  int row0 = blockIdx.x * 128 + wave * 16;
  const _Float16* abase = f60 + (size_t)(row0 + (lane & 15)) * 64;
  v16h a0 = load_afrag(abase, lane, 0);
  v16h a1 = load_afrag(abase, lane, 1);
  float wm[8];
  for (int i = 0; i < 8; ++i) {
    int r = row0 + i + ((lane >> 4) << 3);
    wm[i] = maskf[r / KNBk];
  }
  for (int ct = 0; ct < 8; ++ct) {
    int nn = (ct << 4) + (lane & 15);
    v16h bb0 = load_bfrag(Bl, 64, nn, lane, 0);
    v16h bb1 = load_bfrag(Bl, 64, nn, lane, 1);
    v8f acc = {};
    acc = wmma16(a0, bb0, acc);
    acc = wmma16(a1, bb1, acc);
    float bias = (nn < 120) ? b8[nn] : 0.f;
    float su = 0.f, qu = 0.f;
    for (int i = 0; i < 8; ++i) {
      float v = acc[i] + bias;
      su += wm[i] * v; qu += wm[i] * v * v;
    }
    atomicAdd(&sS[nn], su);
    atomicAdd(&sQ[nn], qu);
  }
  __syncthreads();
  if (t < 128) { atomicAdd(&stats[48 + t], sS[t]); atomicAdd(&stats[176 + t], sQ[t]); }
}

// ---------------- kernel 4: mlp fc1(WMMA)+BN+ReLU+fc2(WMMA) -> att_raw ----
__global__ __launch_bounds__(256) void k_mlp2(
    const _Float16* __restrict__ f60,
    const float* __restrict__ w7, const float* __restrict__ b8,
    const float* __restrict__ par,
    const float* __restrict__ w11, const float* __restrict__ b12,
    float* __restrict__ att_raw) {
  __shared__ __align__(32) _Float16 Bl[128 * 64];   // fc1 weights f16, col-major-K
  __shared__ __align__(32) _Float16 B2[16 * 128];   // fc2 weights f16
  __shared__ __align__(32) _Float16 At[8 * 16 * 128]; // BN'd activations per wave
  int t = threadIdx.x;
  for (int i = t; i < 128 * 64; i += 256) {
    int nn = i >> 6, kk = i & 63;
    Bl[i] = (_Float16)((nn < 120 && kk < 60) ? w7[nn * 60 + kk] : 0.f);
  }
  for (int i = t; i < 16 * 128; i += 256) {
    int nn = i >> 7, kk = i & 127;
    B2[i] = (_Float16)((nn < 5 && kk < 120) ? w11[nn * 120 + kk] : 0.f);
  }
  __syncthreads();

  int wave = t >> 5, lane = t & 31;
  int row0 = blockIdx.x * 128 + wave * 16;
  const _Float16* abase = f60 + (size_t)(row0 + (lane & 15)) * 64;
  v16h a0 = load_afrag(abase, lane, 0);
  v16h a1 = load_afrag(abase, lane, 1);
  _Float16* Aw = At + wave * 16 * 128;
  for (int ct = 0; ct < 8; ++ct) {
    int nn = (ct << 4) + (lane & 15);
    v16h bb0 = load_bfrag(Bl, 64, nn, lane, 0);
    v16h bb1 = load_bfrag(Bl, 64, nn, lane, 1);
    v8f acc = {};
    acc = wmma16(a0, bb0, acc);
    acc = wmma16(a1, bb1, acc);
    float bias = (nn < 120) ? b8[nn] : 0.f;
    float sc = (nn < 120) ? par[32 + nn] : 0.f;
    float sh = (nn < 120) ? par[160 + nn] : 0.f;
    for (int i = 0; i < 8; ++i) {
      int M = i + ((lane >> 4) << 3);
      float v = acc[i] + bias;
      float z = (nn < 120) ? fmaxf(v * sc + sh, 0.f) : 0.f;
      Aw[M * 128 + nn] = (_Float16)z;
    }
  }
  __syncthreads();
  // fc2: K=128 (4 k-steps), one 16-col tile (5 valid)
  const _Float16* arow2 = Aw + (lane & 15) * 128;
  v8f acc2 = {};
  for (int ks = 0; ks < 4; ++ks) {
    v16h aa = load_afrag(arow2, lane, ks);
    v16h bb = load_bfrag(B2, 128, lane & 15, lane, ks);
    acc2 = wmma16(aa, bb, acc2);
  }
  int col = lane & 15;
  if (col < 5) {
    for (int i = 0; i < 8; ++i) {
      int M = i + ((lane >> 4) << 3);
      att_raw[(size_t)(row0 + M) * 5 + col] = acc2[i] + b12[col];
    }
  }
}

// ---------------- kernel 5: softmax + coords*att + post fc1 + stats -------
__global__ __launch_bounds__(256) void k_post1(
    const float* __restrict__ att_raw, const float* __restrict__ coords,
    const float* __restrict__ maskf,
    const float* __restrict__ w13, const float* __restrict__ b14,
    float* __restrict__ p1, float* __restrict__ stats) {
  __shared__ float sS[5], sQ[5];
  int t = threadIdx.x;
  if (t < 5) { sS[t] = 0.f; sQ[t] = 0.f; }
  __syncthreads();
  int n = blockIdx.x * 256 + t;
  float a[CCH][KNBk];
  for (int k = 0; k < KNBk; ++k)
    for (int ch = 0; ch < CCH; ++ch)
      a[ch][k] = att_raw[(size_t)(n * KNBk + k) * 5 + ch];
  float v25[25];
  for (int ch = 0; ch < CCH; ++ch) {
    float mxv = -1e30f;
    for (int k = 0; k < KNBk; ++k) mxv = fmaxf(mxv, a[ch][k]);
    float e[KNBk], s = 0.f;
    for (int k = 0; k < KNBk; ++k) { e[k] = __expf(a[ch][k] - mxv); s += e[k]; }
    for (int k = 0; k < KNBk; ++k)
      v25[ch * KNBk + k] = coords[(size_t)n * 25 + ch * KNBk + k] * (e[k] / s);
  }
  float m = maskf[n];
  float p[5];
  for (int o = 0; o < 5; ++o) {
    float acc = b14[o];
    for (int j = 0; j < 25; ++j) acc += w13[o * 25 + j] * v25[j];
    p[o] = acc;
    p1[(size_t)n * 5 + o] = acc;
  }
  if (m > 0.f)
    for (int o = 0; o < 5; ++o) { atomicAdd(&sS[o], p[o]); atomicAdd(&sQ[o], p[o] * p[o]); }
  __syncthreads();
  if (t < 5) { atomicAdd(&stats[304 + t], sS[t]); atomicAdd(&stats[320 + t], sQ[t]); }
}

// ---------------- kernel 6: post BN+ReLU+fc2, masked scatter --------------
__global__ __launch_bounds__(256) void k_post2(
    const float* __restrict__ p1, const float* __restrict__ maskf,
    const float* __restrict__ par,
    const float* __restrict__ w17, const float* __restrict__ b18,
    float* __restrict__ out) {
  int n = blockIdx.x * 256 + threadIdx.x;
  float m = maskf[n];
  float z[5];
  for (int o = 0; o < 5; ++o) {
    float v = p1[(size_t)n * 5 + o] * par[288 + o] + par[304 + o];
    z[o] = fmaxf(v, 0.f);
  }
  int bi = n >> 16, rem = n & 65535, y = rem >> 8, x = rem & 255;
  for (int o = 0; o < 5; ++o) {
    float acc = b18[o];
    for (int i = 0; i < 5; ++i) acc += w17[o * 5 + i] * z[i];
    out[(((size_t)bi * 5 + o) * HH + y) * WW + x] = (m > 0.f) ? acc : 0.f;
  }
}

extern "C" void kernel_launch(void* const* d_in, const int* in_sizes, int n_in,
                              void* d_out, int out_size, void* d_ws, size_t ws_size,
                              hipStream_t stream) {
  const float* in      = (const float*)d_in[0];
  const float* vfe_w1  = (const float*)d_in[1];
  const float* vfe_b1  = (const float*)d_in[2];
  const float* vfe_g   = (const float*)d_in[3];
  const float* vfe_b   = (const float*)d_in[4];
  const float* vfe_w2  = (const float*)d_in[5];
  const float* vfe_b2  = (const float*)d_in[6];
  const float* mlp_w1  = (const float*)d_in[7];
  const float* mlp_b1  = (const float*)d_in[8];
  const float* mlp_g   = (const float*)d_in[9];
  const float* mlp_b   = (const float*)d_in[10];
  const float* mlp_w2  = (const float*)d_in[11];
  const float* mlp_b2  = (const float*)d_in[12];
  const float* post_w1 = (const float*)d_in[13];
  const float* post_b1 = (const float*)d_in[14];
  const float* post_g  = (const float*)d_in[15];
  const float* post_b  = (const float*)d_in[16];
  const float* post_w2 = (const float*)d_in[17];
  const float* post_b2 = (const float*)d_in[18];
  float* out = (float*)d_out;

  char* ws = (char*)d_ws;
  size_t off = 0;
  auto grab = [&](size_t bytes) { size_t r = off; off = (off + bytes + 255) & ~(size_t)255; return r; };
  float*    coords = (float*)   (ws + grab((size_t)NPIX * 25 * 4));
  float*    maskf  = (float*)   (ws + grab((size_t)NPIX * 4));
  _Float16* f60    = (_Float16*)(ws + grab((size_t)NKROW * 64 * 2));
  float*    attraw = (float*)   (ws + grab((size_t)NKROW * 5 * 4));
  float*    p1     = (float*)   (ws + grab((size_t)NPIX * 5 * 4));
  float*    stats  = (float*)   (ws + grab(512 * 4));
  float*    par    = (float*)   (ws + grab(512 * 4));

  dim3 blk(256);
  // 0) zero the accumulators (must happen every call for graph replay)
  k_zero<<<2, blk, 0, stream>>>(stats, 512);
  // 1) unfold/select/gather + vfe fc1 stats
  k_prep<<<NPIX / 256, blk, 0, stream>>>(in, vfe_w1, vfe_b1, coords, maskf, stats);
  // 2) vfe BN finalize (count = sum(m)*KNB)
  k_finalize<<<1, 128, 0, stream>>>(stats, par, vfe_g, vfe_b, 16, 32, 0, 16, 10, (float)KNBk);
  // 3) vfe apply + max/mean -> f60
  k_vfe2<<<NPIX / 256, blk, 0, stream>>>(coords, vfe_w1, vfe_b1, par, vfe_w2, vfe_b2, f60);
  // 4) mlp fc1 via WMMA, accumulate BN stats
  k_mlp_stats<<<NKROW / 128, blk, 0, stream>>>(f60, maskf, mlp_w1, mlp_b1, stats);
  // 5) mlp BN finalize
  k_finalize<<<1, 128, 0, stream>>>(stats, par, mlp_g, mlp_b, 48, 176, 32, 160, 120, (float)KNBk);
  // 6) mlp fc1+BN+ReLU+fc2 via WMMA -> attention logits
  k_mlp2<<<NKROW / 128, blk, 0, stream>>>(f60, mlp_w1, mlp_b1, par, mlp_w2, mlp_b2, attraw);
  // 7) softmax + weighted coords + post fc1 + stats
  k_post1<<<NPIX / 256, blk, 0, stream>>>(attraw, coords, maskf, post_w1, post_b1, p1, stats);
  // 8) post BN finalize (count = sum(m))
  k_finalize<<<1, 128, 0, stream>>>(stats, par, post_g, post_b, 304, 320, 288, 304, 5, 1.0f);
  // 9) post BN+ReLU+fc2 + masked scatter to [b,c,h,w]
  k_post2<<<NPIX / 256, blk, 0, stream>>>(p1, maskf, par, post_w2, post_b2, out);
}
